// VectorQuantizer_35802847379920
// MI455X (gfx1250) — matrix operand order repaired and checked by reading, not compiled, not run on hardware
//
#include <hip/hip_runtime.h>
#include <hip/hip_bf16.h>

// ---------------------------------------------------------------------------
// VQ-VAE vector quantizer for MI455X (gfx1250, wave32, WMMA + async→LDS)
//   z_e:   [32, 256, 32, 32] f32   (d_in[0])
//   emb_w: [8192, 256]       f32   (d_in[1])
//   out:   z_q_st (8388608 f32) | vq_loss (1) | perplexity (1) | indices (32768)
// ---------------------------------------------------------------------------

typedef __attribute__((ext_vector_type(16))) __bf16 v16bf;
typedef __attribute__((ext_vector_type(8)))  __bf16 v8bf;
typedef __attribute__((ext_vector_type(8)))  float  v8f;

#define NUM_EMB   8192
#define EMB_DIM   256
#define N_TOK     32768
#define OUT_ZQ    8388608
#define TOK_PER_WG 128      // 8 waves * 16 tokens
#define N_TILES   (NUM_EMB / 16)

union AU { v16bf v; v8bf h[2]; };

__device__ __forceinline__ float block_reduce_sum(float v) {
  __shared__ float sred[8];
  const int lane = threadIdx.x & 31;
  const int wid  = threadIdx.x >> 5;
#pragma unroll
  for (int o = 16; o > 0; o >>= 1) v += __shfl_down(v, o, 32);
  if (lane == 0) sred[wid] = v;
  __syncthreads();
  v = (threadIdx.x < (blockDim.x >> 5)) ? sred[threadIdx.x] : 0.0f;
  if (wid == 0) {
#pragma unroll
    for (int o = 4; o > 0; o >>= 1) v += __shfl_down(v, o, 32);
  }
  return v;  // valid in thread 0
}

// Async global->LDS copy of 32 bytes per thread (2 x b128, ASYNCcnt-tracked).
// INST_OFFSET is added to BOTH the LDS and global addresses (ISA 08 §4.4),
// so one address pair serves both halves.
__device__ __forceinline__ void async_copy32(unsigned lds_addr, const void* gaddr) {
  asm volatile("global_load_async_to_lds_b128 %0, %1, off\n\t"
               "global_load_async_to_lds_b128 %0, %1, off offset:16"
               :: "v"(lds_addr), "v"(gaddr) : "memory");
}

// ---- 1) emb f32 -> bf16 (row-major) + ||e||^2 per code ---------------------
__global__ __launch_bounds__(256) void vq_prep_emb(const float* __restrict__ emb,
                                                   __bf16* __restrict__ embb,
                                                   float* __restrict__ enorm) {
  const int r = blockIdx.x;          // code row
  const int c = threadIdx.x;         // channel
  const float v = emb[(size_t)r * EMB_DIM + c];
  embb[(size_t)r * EMB_DIM + c] = (__bf16)v;
  const float s = block_reduce_sum(v * v);
  if (threadIdx.x == 0) enorm[r] = s;
}

// ---- zero the stats scratch ------------------------------------------------
__global__ __launch_bounds__(256) void vq_init(float* __restrict__ counts,
                                               float* __restrict__ msea) {
  const int i = blockIdx.x * 256 + threadIdx.x;
  if (i < NUM_EMB) counts[i] = 0.0f;
  if (i == 0) msea[0] = 0.0f;
}

// ---- 2) fused distance GEMM (bf16 WMMA) + argmin ---------------------------
// Each wave: 16 tokens; B operand (16 tok x 256 k bf16) resident in 64 VGPRs.
// Workgroup sweeps all 512 code tiles with DOUBLE-BUFFERED LDS filled by
// GLOBAL_LOAD_ASYNC_TO_LDS_B128 (copy of tile i+1 overlaps WMMAs of tile i).
// Score = ||e||^2 - 2*(e . z): ||z||^2 is row-constant, dropped for argmin.
__global__ __launch_bounds__(256, 1) void vq_argmin(const float* __restrict__ z,
                                                    const __bf16* __restrict__ embb,
                                                    const float* __restrict__ enorm,
                                                    int* __restrict__ idx_out) {
  __shared__ __align__(16) __bf16 sA[2][16 * EMB_DIM];  // 2 x 8 KB emb tiles
  __shared__ __align__(16) float  sEn[NUM_EMB];         // 32 KB code norms

  const int tid  = threadIdx.x;
  const int lane = tid & 31;
  const int wave = tid >> 5;
  const int m    = lane & 15;          // code row within tile / token col
  const int hi8  = (lane >> 4) << 3;   // 0 or 8  (C/D + A half-select)
  const int hi16 = hi8 << 1;           // 0 or 16 (unused)
  const int tokenBase = blockIdx.x * TOK_PER_WG + wave * 16;

  // Cooperative tile-copy mapping: 256 threads x 32B = 8 KB tile.
  const int trow = tid >> 4;          // 0..15 code row in tile
  const int tcol = (tid & 15) * 16;   // element offset (16 bf16 = 32 B)
  const __bf16* gsrc = embb + (size_t)trow * EMB_DIM + tcol;
  unsigned ldsDst[2];
  ldsDst[0] = (unsigned)(uintptr_t)&sA[0][trow * EMB_DIM + tcol];
  ldsDst[1] = (unsigned)(uintptr_t)&sA[1][trow * EMB_DIM + tcol];

  // Kick off the async copy of tile 0, then hide z/enorm staging under it.
  async_copy32(ldsDst[0], gsrc);

  // Stage ||e||^2 into LDS once.
  for (int i = tid; i < NUM_EMB; i += 256) sEn[i] = enorm[i];

  // Load B operand (this wave's 16 tokens, all 256 k) into registers as bf16.
  // Dense B 32x16 layout: lane holds column N = lane%16, K {hi8..hi8+7} and
  // {hi8+16..hi8+23} (matching A), packed 2/VGPR.
  const int token = tokenBase + m;
  const float* __restrict__ zcol =
      z + ((size_t)(token >> 10) << 18) + (size_t)(token & 1023);
  v16bf Breg[8];
#pragma unroll
  for (int kb = 0; kb < 8; ++kb) {
#pragma unroll
    for (int j = 0; j < 16; ++j) {
      const int k = kb * 32 + hi8 + j + (j & 8);
      Breg[kb][j] = (__bf16)zcol[(size_t)k << 10];
    }
  }

  float best = 3.4e38f;
  int   bidx = 0;

  for (int ti = 0; ti < N_TILES; ++ti) {
    const int tb = ti << 4;
    // Issue async copy of the NEXT tile into the other buffer, then retire
    // this tile's two ops (in-order completion => wait asynccnt <= 2).
    if (ti + 1 < N_TILES) {
      async_copy32(ldsDst[(ti + 1) & 1],
                   gsrc + (size_t)(ti + 1) * (16 * EMB_DIM));
      asm volatile("s_wait_asynccnt 0x2" ::: "memory");
    } else {
      asm volatile("s_wait_asynccnt 0x0" ::: "memory");
    }
    __syncthreads();                      // all waves' copies of tile ti landed

    const __bf16* __restrict__ tile = &sA[ti & 1][0];

    // Preload all 8 A chunks so the 16 ds_load_b128 pipeline and the 8 WMMAs
    // issue back-to-back (no per-WMMA s_wait_dscnt 0).
    AU a[8];
#pragma unroll
    for (int kb = 0; kb < 8; ++kb) {
      const int off = kb * 32 + hi8;
      // 16-bit A layout: lane half picks K {off..off+7} and {off+16..off+23}
      a[kb].h[0] = *(const v8bf*)&tile[m * EMB_DIM + off];
      a[kb].h[1] = *(const v8bf*)&tile[m * EMB_DIM + off + 16];
    }
    const float4 e0 = *(const float4*)&sEn[tb + hi8];
    const float4 e1 = *(const float4*)&sEn[tb + hi8 + 4];

    v8f acc = {};
#pragma unroll
    for (int kb = 0; kb < 8; ++kb)
      acc = __builtin_amdgcn_wmma_f32_16x16x32_bf16(
          false, a[kb].v, false, Breg[kb], (short)0, acc, false, false);

    // Fused argmin epilogue. C/D layout: VGPR r -> code (tb + hi8 + r),
    // lane%16 -> token. 8 scores per lane, same token.
    const float en[8] = {e0.x, e0.y, e0.z, e0.w, e1.x, e1.y, e1.z, e1.w};
#pragma unroll
    for (int r = 0; r < 8; ++r) {
      const float sc = fmaf(-2.0f, acc[r], en[r]);
      const int code = tb + hi8 + r;
      if (sc < best) { best = sc; bidx = code; }
    }
    __syncthreads();   // buffer may be re-targeted by next iteration's issue
  }

  // Combine the two lane halves tracking the same token.
  const float ob = __shfl_xor(best, 16, 32);
  const int   oi = __shfl_xor(bidx, 16, 32);
  if (ob < best || (ob == best && oi < bidx)) { best = ob; bidx = oi; }
  if (lane < 16) idx_out[tokenBase + lane] = bidx;
}

// ---- 3) gather z_q (f32 codebook), scatter to NCHW, MSE + histogram --------
__global__ __launch_bounds__(256) void vq_gather(const float* __restrict__ z,
                                                 const float* __restrict__ emb,
                                                 const int* __restrict__ idx,
                                                 float* __restrict__ out,
                                                 float* __restrict__ msea,
                                                 float* __restrict__ counts,
                                                 float* __restrict__ out_idx) {
  const int t = blockIdx.x;          // token
  const int c = threadIdx.x;         // channel
  const int code = idx[t];
  const int b  = t >> 10;
  const int hw = t & 1023;
  const float e = emb[(size_t)code * EMB_DIM + c];
  const size_t off = ((size_t)(b * EMB_DIM + c) << 10) + hw;
  out[off] = e;                      // straight-through output == z_q numerically
  const float d = z[off] - e;
  const float s = block_reduce_sum(d * d);
  if (c == 0) {
    atomicAdd(msea, s);
    atomicAdd(&counts[code], 1.0f);
    out_idx[t] = (float)code;
  }
}

// ---- 4) perplexity + vq_loss scalars ---------------------------------------
__global__ __launch_bounds__(256) void vq_finalize(const float* __restrict__ counts,
                                                   const float* __restrict__ msea,
                                                   float* __restrict__ out_loss,
                                                   float* __restrict__ out_ppl) {
  float local = 0.0f;
  for (int i = threadIdx.x; i < NUM_EMB; i += 256) {
    const float cnt = counts[i];
    if (cnt > 0.0f) {
      const float p = cnt * (1.0f / (float)N_TOK);
      local += p * logf(p);
    }
  }
  const float s = block_reduce_sum(local);
  if (threadIdx.x == 0) {
    out_ppl[0]  = expf(-s);
    out_loss[0] = 1.25f * (msea[0] / (float)OUT_ZQ);
  }
}

// ---------------------------------------------------------------------------
extern "C" void kernel_launch(void* const* d_in, const int* in_sizes, int n_in,
                              void* d_out, int out_size, void* d_ws, size_t ws_size,
                              hipStream_t stream) {
  (void)in_sizes; (void)n_in; (void)out_size; (void)ws_size;
  const float* z   = (const float*)d_in[0];   // [32,256,32,32]
  const float* emb = (const float*)d_in[1];   // [8192,256]

  char* ws = (char*)d_ws;
  __bf16* embb  = (__bf16*)ws;                                   // 4 MB
  float*  enorm = (float*)(ws + (4u << 20));                     // 32 KB
  int*    idxb  = (int*)(ws + (4u << 20) + (32u << 10));         // 128 KB
  float*  counts= (float*)(ws + (4u << 20) + (160u << 10));      // 32 KB
  float*  msea  = (float*)(ws + (4u << 20) + (192u << 10));      // 4 B

  float* out      = (float*)d_out;
  float* out_loss = out + OUT_ZQ;
  float* out_ppl  = out + OUT_ZQ + 1;
  float* out_idx  = out + OUT_ZQ + 2;

  vq_prep_emb<<<NUM_EMB, 256, 0, stream>>>(emb, embb, enorm);
  vq_init<<<(NUM_EMB + 255) / 256, 256, 0, stream>>>(counts, msea);
  vq_argmin<<<N_TOK / TOK_PER_WG, 256, 0, stream>>>(z, embb, enorm, idxb);
  vq_gather<<<N_TOK, 256, 0, stream>>>(z, emb, idxb, out, msea, counts, out_idx);
  vq_finalize<<<1, 256, 0, stream>>>(counts, msea, out_loss, out_ppl);
}
